// GraphAttentionLayer_16045997818087
// MI455X (gfx1250) — matrix operand order repaired and checked by reading, not compiled, not run on hardware
//
#include <hip/hip_runtime.h>

typedef __attribute__((ext_vector_type(2)))  float  v2f;
typedef __attribute__((ext_vector_type(8)))  float  v8f;
typedef __attribute__((ext_vector_type(16))) __bf16 v16bf;
typedef __attribute__((ext_vector_type(4)))  unsigned int u32x4;
typedef __attribute__((ext_vector_type(8)))  int i32x8;
typedef __attribute__((ext_vector_type(4)))  int i32x4;

#define N_NODES 2048
#define IN_F    512
#define OUT_F   512
#define N_HEADS 8
#define N_HID   64
#define NEG_SLOPE 0.2f

__device__ __forceinline__ v8f wmma_f32(v2f a, v2f b, v8f c) {
    return __builtin_amdgcn_wmma_f32_16x16x4_f32(false, a, false, b, (short)0, c, false, false);
}
__device__ __forceinline__ v8f wmma_bf16(v16bf a, v16bf b, v8f c) {
    return __builtin_amdgcn_wmma_f32_16x16x32_bf16(false, a, false, b, (short)0, c, false, false);
}

// ---------------- Kernel 1: g = h @ W^T (fp32 WMMA, exact) -------------------------
// TDM stages the shared 16x512 h tile into LDS once per block; 4 waves then run
// the K loop with A-operands from LDS and per-wave W rows from global (L2).
__global__ void gat_gemm_g(const float* __restrict__ h,
                           const float* __restrict__ W,
                           float* __restrict__ g,
                           __bf16* __restrict__ gbT) {
    __shared__ float hTile[16 * IN_F];                 // 32 KB, LDS offset 0

    const int lane  = threadIdx.x & 31;
    const int wave  = threadIdx.x >> 5;
    const int lm    = lane & 15;
    const int hh    = lane >> 4;
    const int mBase = blockIdx.x * 16;
    const int nBase = blockIdx.y * 256 + wave * 64;

    if (wave == 0) {
        // Tensor DMA descriptor (D#): 2-D tile, 16 rows x 512 cols of fp32
        const unsigned long long gaddr =
            (unsigned long long)(const void*)(h + (size_t)mBase * IN_F);
        u32x4 g0;
        g0[0] = 1u;                                    // count=1 (valid), user mode
        g0[1] = 0u;                                    // lds_addr = 0 (hTile)
        g0[2] = (unsigned)(gaddr & 0xFFFFFFFFu);       // global_addr[31:0]
        g0[3] = (unsigned)((gaddr >> 32) & 0x01FFFFFFu) | (2u << 30); // addr[56:32], type=2

        i32x8 g1;
        g1[0] = (int)(2u << 16);                       // workgroup_mask=0, data_size=2 (4B)
        g1[1] = (int)((IN_F & 0xFFFFu) << 16);         // tensor_dim0[15:0] in bits 63:48
        g1[2] = (int)((IN_F >> 16) | ((N_NODES & 0xFFFFu) << 16)); // dim0 hi | dim1 lo
        g1[3] = (int)((N_NODES >> 16) | ((unsigned)IN_F << 16));   // dim1 hi | tile_dim0=512
        g1[4] = 16;                                    // tile_dim1=16, tile_dim2=0
        g1[5] = IN_F;                                  // tensor_dim0_stride = 512
        g1[6] = 0;
        g1[7] = 0;

        i32x4 z4 = {0, 0, 0, 0};
        i32x8 z8 = {0, 0, 0, 0, 0, 0, 0, 0};
        __builtin_amdgcn_tensor_load_to_lds(g0, g1, z4, z4, z8, 0);
        __builtin_amdgcn_s_wait_tensorcnt(0);
    }
    __syncthreads();

    v8f c0 = {}, c1 = {}, c2 = {}, c3 = {};

    const float* hRow = &hTile[lm * IN_F + 2 * hh];    // LDS (ds_load_b64)
    const float* w0 = W + (size_t)(nBase +  0 + lm) * IN_F + 2 * hh;
    const float* w1 = W + (size_t)(nBase + 16 + lm) * IN_F + 2 * hh;
    const float* w2 = W + (size_t)(nBase + 32 + lm) * IN_F + 2 * hh;
    const float* w3 = W + (size_t)(nBase + 48 + lm) * IN_F + 2 * hh;

    for (int k = 0; k < IN_F; k += 4) {
        v2f a  = *(const v2f*)(hRow + k);
        v2f b0 = *(const v2f*)(w0 + k);
        v2f b1 = *(const v2f*)(w1 + k);
        v2f b2 = *(const v2f*)(w2 + k);
        v2f b3 = *(const v2f*)(w3 + k);
        c0 = wmma_f32(a, b0, c0);
        c1 = wmma_f32(a, b1, c1);
        c2 = wmma_f32(a, b2, c2);
        c3 = wmma_f32(a, b3, c3);
    }

    // C layout: lane holds N=lm, VGPR r holds M = r + 8*hh
    #pragma unroll
    for (int r = 0; r < 8; ++r) {
        const size_t m = (size_t)(mBase + r + 8 * hh);
        float* o = g + m * OUT_F + nBase + lm;
        o[ 0] = c0[r];
        o[16] = c1[r];
        o[32] = c2[r];
        o[48] = c3[r];
        // transposed bf16 copy: gbT[f][node]
        gbT[(size_t)(nBase +  0 + lm) * N_NODES + m] = (__bf16)c0[r];
        gbT[(size_t)(nBase + 16 + lm) * N_NODES + m] = (__bf16)c1[r];
        gbT[(size_t)(nBase + 32 + lm) * N_NODES + m] = (__bf16)c2[r];
        gbT[(size_t)(nBase + 48 + lm) * N_NODES + m] = (__bf16)c3[r];
    }
}

// ---------------- Kernel 2: elT/erT = per-head dot(g, a_l/a_r), head-major ---------
__global__ void gat_edge(const float* __restrict__ g,
                         const float* __restrict__ attn_w,
                         float* __restrict__ elT,
                         float* __restrict__ erT) {
    const int idx = blockIdx.x * blockDim.x + threadIdx.x;   // 0 .. 2048*8-1
    const int n  = idx >> 3;
    const int hd = idx & 7;
    const float* gr = g + (size_t)n * OUT_F + hd * N_HID;
    float sl = 0.f, sr = 0.f;
    #pragma unroll 8
    for (int f = 0; f < N_HID; ++f) {
        const float v = gr[f];
        sl += v * attn_w[f];
        sr += v * attn_w[N_HID + f];
    }
    elT[hd * N_NODES + n] = sl;
    erT[hd * N_NODES + n] = sr;
}

// ---------------- Kernel 3: fused masked-softmax + aggregation (bf16 WMMA) ---------
// One wave per (16-row block, head). Online softmax over j in tiles of 32;
// aggregation acc += P(16x32,bf16) @ G(32x64,bf16) with fp32 accumulate.
__global__ void gat_attn(const __bf16* __restrict__ gbT,
                         const int*    __restrict__ adj,
                         const float*  __restrict__ elT,
                         const float*  __restrict__ erT,
                         float* __restrict__ out) {
    const int lane  = threadIdx.x & 31;
    const int lm    = lane & 15;
    const int hh    = lane >> 4;
    const int iBase = blockIdx.x * 16;
    const int head  = blockIdx.y;

    const int row   = iBase + lm;                      // softmax row for this lane
    const float eli = elT[head * N_NODES + row];
    const int*   adjRow = adj + (size_t)row * N_NODES;
    const float* erH    = erT + (size_t)head * N_NODES;

    // B columns this lane streams (gbT rows), f = head*64 + chunk*16 + lm
    const __bf16* gc0 = gbT + (size_t)(head * N_HID +  0 + lm) * N_NODES + 16 * hh;
    const __bf16* gc1 = gbT + (size_t)(head * N_HID + 16 + lm) * N_NODES + 16 * hh;
    const __bf16* gc2 = gbT + (size_t)(head * N_HID + 32 + lm) * N_NODES + 16 * hh;
    const __bf16* gc3 = gbT + (size_t)(head * N_HID + 48 + lm) * N_NODES + 16 * hh;

    float m = -1e30f;                                  // running max (lanes lm, lm+16)
    float s = 0.f;                                     // running sumexp
    v8f c0 = {}, c1 = {}, c2 = {}, c3 = {};

    for (int jb = 0; jb < N_NODES; jb += 32) {
        // A-layout (bf16 16x32): element e -> K = 16*hh+e (matches B)
        const int j0 = jb + 16 * hh;
        const int j1 = jb + 16 * hh + 8;

        int   av[16];
        float ev[16];
        ((int4*)av)[0] = *(const int4*)(adjRow + j0);
        ((int4*)av)[1] = *(const int4*)(adjRow + j0 + 4);
        ((int4*)av)[2] = *(const int4*)(adjRow + j1);
        ((int4*)av)[3] = *(const int4*)(adjRow + j1 + 4);
        ((float4*)ev)[0] = *(const float4*)(erH + j0);
        ((float4*)ev)[1] = *(const float4*)(erH + j0 + 4);
        ((float4*)ev)[2] = *(const float4*)(erH + j1);
        ((float4*)ev)[3] = *(const float4*)(erH + j1 + 4);

        float p[16];
        float mt = -1e30f;
        #pragma unroll
        for (int q = 0; q < 16; ++q) {
            float e = eli + ev[q];
            e = (e >= 0.f) ? e : NEG_SLOPE * e;
            e = (av[q] != 0) ? e : -1e30f;
            p[q] = e;
            mt = fmaxf(mt, e);
        }
        // lanes lm / lm+16 together cover all 32 j's of this row
        mt = fmaxf(mt, __shfl_xor(mt, 16, 32));
        const float mNew  = fmaxf(m, mt);
        const float scale = (m > -1e29f) ? __expf(m - mNew) : 0.f;

        float ts = 0.f;
        #pragma unroll
        for (int q = 0; q < 16; ++q) {
            const float pv = (p[q] > -1e29f) ? __expf(p[q] - mNew) : 0.f;
            p[q] = pv;
            ts += pv;
        }
        ts += __shfl_xor(ts, 16, 32);
        s = s * scale + ts;
        m = mNew;

        // rescale accumulators; lane needs scale of rows r + 8*hh
        float sc[8];
        #pragma unroll
        for (int r = 0; r < 8; ++r)
            sc[r] = __shfl(scale, (r + 8 * hh) & 15, 32);
        #pragma unroll
        for (int r = 0; r < 8; ++r) {
            c0[r] *= sc[r]; c1[r] *= sc[r]; c2[r] *= sc[r]; c3[r] *= sc[r];
        }

        // pack probabilities -> bf16 A operand
        v16bf a;
        #pragma unroll
        for (int q = 0; q < 16; ++q) a[q] = (__bf16)p[q];

        // B operands: 16 consecutive bf16 per lane (32B contiguous loads)
        v16bf b0 = *(const v16bf*)(gc0 + jb);
        v16bf b1 = *(const v16bf*)(gc1 + jb);
        v16bf b2 = *(const v16bf*)(gc2 + jb);
        v16bf b3 = *(const v16bf*)(gc3 + jb);
        __builtin_prefetch(gc0 + jb + 32, 0, 0);
        __builtin_prefetch(gc1 + jb + 32, 0, 0);

        c0 = wmma_bf16(a, b0, c0);
        c1 = wmma_bf16(a, b1, c1);
        c2 = wmma_bf16(a, b2, c2);
        c3 = wmma_bf16(a, b3, c3);
    }

    // normalize and store
    const float inv = (s > 0.f) ? (1.f / s) : 0.f;     // lanes lm, lm+16 agree
    float invr[8];
    #pragma unroll
    for (int r = 0; r < 8; ++r)
        invr[r] = __shfl(inv, (r + 8 * hh) & 15, 32);
    #pragma unroll
    for (int r = 0; r < 8; ++r) {
        const size_t mrow = (size_t)(iBase + r + 8 * hh);
        float* o = out + mrow * OUT_F + head * N_HID + lm;
        o[ 0] = c0[r] * invr[r];
        o[16] = c1[r] * invr[r];
        o[32] = c2[r] * invr[r];
        o[48] = c3[r] * invr[r];
    }
}

extern "C" void kernel_launch(void* const* d_in, const int* in_sizes, int n_in,
                              void* d_out, int out_size, void* d_ws, size_t ws_size,
                              hipStream_t stream) {
    const float* h   = (const float*)d_in[0];
    const int*   adj = (const int*)d_in[1];
    const float* W   = (const float*)d_in[2];
    const float* aw  = (const float*)d_in[3];

    char* ws = (char*)d_ws;
    float*  g   = (float*)ws;                                   // 4 MB
    __bf16* gbT = (__bf16*)(ws + (size_t)N_NODES * OUT_F * 4);  // 2 MB
    float*  elT = (float*)(ws + (size_t)N_NODES * OUT_F * 6);   // 64 KB
    float*  erT = elT + (size_t)N_NODES * N_HEADS;              // 64 KB

    // 1) g = h @ W^T  (TDM-staged h tile + fp32 WMMA, + bf16 transposed copy)
    dim3 grid1(N_NODES / 16, 2);
    gat_gemm_g<<<grid1, 128, 0, stream>>>(h, W, g, gbT);

    // 2) elT / erT (head-major)
    gat_edge<<<(N_NODES * N_HEADS) / 256, 256, 0, stream>>>(g, aw, elT, erT);

    // 3) fused masked softmax + bf16 WMMA aggregation
    dim3 grid3(N_NODES / 16, N_HEADS);
    gat_attn<<<grid3, 32, 0, stream>>>(gbT, adj, elT, erT, (float*)d_out);
}
